// EncoderDecoder_36266703848054
// MI455X (gfx1250) — matrix operand order repaired and checked by reading, not compiled, not run on hardware
//
#include <hip/hip_runtime.h>
#include <hip/hip_bf16.h>

// ---------------------------------------------------------------------------
// CDNA5 (gfx1250) GRU encoder-decoder.
// All matmuls use v_wmma_f32_16x16x32_bf16 (wave32 matrix cores, fp32 acc).
// Weights are converted+transposed to bf16 Kt x N once per launch (~5.7MB,
// L2-resident in the 192MB L2) so GEMM B-tiles stage into LDS with b128
// loads/stores only. GEMMs are double-buffered in LDS (one barrier per
// K-chunk), each wave register-tiles a 16x32 output (2 WMMA per chunk sharing
// one A fragment), and the hot loop is branch-free for the full-width gate
// GEMMs (guards exist only in the N=96 projection instantiation). Bias /
// residual / mask epilogues are template-fused into the GEMM so each of the
// 100 serial time steps is 7-8 kernel launches.
// ---------------------------------------------------------------------------

typedef __bf16 bf16_t;
typedef bf16_t v16bf __attribute__((ext_vector_type(16)));
typedef bf16_t v8bf  __attribute__((ext_vector_type(8)));
typedef float  v8f   __attribute__((ext_vector_type(8)));

#define T_LEN  50
#define BATCH  256
#define FDIM   96
#define HDIM   512
#define RNN_IN 288   // F + F + PE_DIM
#define G3     1536  // 3*H

// ---------------- positional encoding ----------------
__device__ __forceinline__ float pe_val(int pos, int f) {
  int i2 = f & ~1;  // 2*(f/2)
  float div = __expf((float)i2 * (-9.210340371976184f / 96.0f)); // -ln(10000)/PE_DIM
  float a = (float)pos * div;
  return (f & 1) ? __cosf(a) : __sinf(a);
}

// ---------------- elementwise kernels ----------------
// d[k*N + n] = (bf16) s[n*K + k]   : weight convert + transpose (once/launch)
__global__ void transpose_to_bf16_k(const float* __restrict__ s, bf16_t* __restrict__ d,
                                    int N, int K) {
  int i = blockIdx.x * blockDim.x + threadIdx.x;
  if (i >= N * K) return;
  int k = i / N, n = i % N;
  d[i] = (bf16_t)s[(size_t)n * K + k];
}

__global__ void mask_k(const float* __restrict__ input, float* __restrict__ m) {
  int i = blockIdx.x * blockDim.x + threadIdx.x;
  if (i >= BATCH * FDIM) return;
  float s = 0.f, sq = 0.f;
  for (int t = 0; t < T_LEN; ++t) {
    float v = input[(size_t)t * BATCH * FDIM + i];
    s += v; sq += v * v;
  }
  float var = (sq - s * s / (float)T_LEN) / (float)(T_LEN - 1);  // ddof=1
  m[i] = (sqrtf(fmaxf(var, 0.f)) > 1e-4f) ? 1.f : 0.f;
}

__global__ void build_xen_k(const float* __restrict__ input, bf16_t* __restrict__ xen) {
  int i = blockIdx.x * blockDim.x + threadIdx.x;
  if (i >= T_LEN * BATCH * FDIM) return;
  int t = i / (BATCH * FDIM);
  int r = i % (BATCH * FDIM);
  int b = r / FDIM, f = r % FDIM;
  float inp = input[i];
  float vel = (t == 0) ? 0.f : inp - input[i - BATCH * FDIM];
  bf16_t* x = xen + ((size_t)t * BATCH + b) * RNN_IN;
  x[f]            = (bf16_t)inp;
  x[FDIM + f]     = (bf16_t)vel;
  x[2 * FDIM + f] = (bf16_t)pe_val(t, f);
}

// GRU gate fusion (biases already folded into gi/gh by the GEMM epilogue):
// h = (1-z)*tanh(in + r*hn) + z*h
__global__ void gru_combine_k(const float* __restrict__ gi, const float* __restrict__ gh,
                              float* __restrict__ h, bf16_t* __restrict__ hb) {
  int i = blockIdx.x * blockDim.x + threadIdx.x;
  if (i >= BATCH * HDIM) return;
  int b = i / HDIM, n = i % HDIM;
  const float* gib = gi + (size_t)b * G3;
  const float* ghb = gh + (size_t)b * G3;
  float r   = gib[n] + ghb[n];
  float z   = gib[HDIM + n] + ghb[HDIM + n];
  float in_ = gib[2 * HDIM + n];
  float hn  = ghb[2 * HDIM + n];
  r = 1.f / (1.f + __expf(-r));
  z = 1.f / (1.f + __expf(-z));
  float nn = tanhf(in_ + r * hn);
  float hv = (1.f - z) * nn + z * h[i];
  h[i]  = hv;
  hb[i] = (bf16_t)hv;
}

__global__ void dec_prep0_k(const float* __restrict__ input, bf16_t* __restrict__ xdec) {
  int i = blockIdx.x * blockDim.x + threadIdx.x;
  if (i >= BATCH * FDIM) return;
  int b = i / FDIM, f = i % FDIM;
  float p   = input[i];                                          // input[0]
  float vel = p - input[(size_t)(T_LEN - 1) * BATCH * FDIM + i]; // input[0]-input[-1]
  bf16_t* x = xdec + (size_t)b * RNN_IN;
  x[f]            = (bf16_t)p;
  x[FDIM + f]     = (bf16_t)vel;
  x[2 * FDIM + f] = (bf16_t)pe_val(T_LEN, f);                    // pe_dec[0]
}

// build decoder input for step t+1 from pred_t (vel = 0)
__global__ void dec_buildx_k(const float* __restrict__ pred, bf16_t* __restrict__ xdec, int t) {
  int i = blockIdx.x * blockDim.x + threadIdx.x;
  if (i >= BATCH * FDIM) return;
  int b = i / FDIM, f = i % FDIM;
  bf16_t* x = xdec + (size_t)b * RNN_IN;
  x[f]            = (bf16_t)pred[i];
  x[FDIM + f]     = (bf16_t)0.f;
  x[2 * FDIM + f] = (bf16_t)pe_val(T_LEN + t + 1, f);            // pe_dec[t+1]
}

// ------ WMMA GEMM: C(MxN,f32) = A(MxK,bf16) * Bt(KxN,bf16) [+bias][+res][*mask]
// Block = 8 wave32 waves, macro-tile 32(M) x 128(N), double-buffered LDS,
// K chunks of 32. Each wave: 16x32 output (2 accumulators share 1 A frag).
// FULLN: N % 128 == 0 -> fully branch-free inner loop and epilogue.
// EP: 0 = +bias, 1 = +bias+resid, 2 = (+bias+resid)*mask.
#define LDA_P 40    // 32 + pad (halfs) -> 80B rows
#define LDB_P 136   // 128 + pad (halfs) -> 272B rows

template <bool FULLN, int EP>
__global__ __launch_bounds__(256)
void wmma_gemm_bf16(const bf16_t* __restrict__ A, int lda,
                    const bf16_t* __restrict__ Bt, int ldb,   // ldb == N
                    float* __restrict__ C, int ldc,
                    int N, int K,
                    const float* __restrict__ bias,    // len N
                    const float* __restrict__ resid,   // MxN (ld=ldc) for EP>=1
                    const float* __restrict__ mask) {  // MxN (ld=ldc) for EP==2
  __shared__ __align__(16) bf16_t sA[2][32 * LDA_P];
  __shared__ __align__(16) bf16_t sB[2][32 * LDB_P];

  const int tid  = threadIdx.x;
  const int lane = tid & 31;
  const int wave = tid >> 5;
  const int m0b  = blockIdx.y * 32;
  const int n0b  = blockIdx.x * 128;
  const int mt   = wave >> 2;          // 0..1
  const int nt   = wave & 3;           // 0..3 -> n pair base nt*32

  // cooperative-load coordinates
  const int ar = tid >> 3;             // 0..31 : A row in tile
  const int ac = (tid & 7) * 4;        // 4 halfs along k
  const int bk = tid >> 3;             // 0..31 : Bt row (k) in tile
  const int bn = (tid & 7) * 16;       // 16 halfs along n

  // A-fragment addressing (16-bit A 16x32 layout):
  // lanes 0-15 hold K {0..7,16..23}; lanes 16-31 hold K {8..15,24..31}
  const int arow  = mt * 16 + (lane & 15);
  const int kbase = (lane & 16) ? 8 : 0;

  const int nk = K / 32;
  const bool bok0 = FULLN || (n0b + bn + 8)  <= N;
  const bool bok1 = FULLN || (n0b + bn + 16) <= N;

  // running global pointers, stepped by constant strides per K-chunk
  const bf16_t* aPtr = A + (size_t)(m0b + ar) * lda + ac;
  const bf16_t* bPtr = Bt + (size_t)bk * ldb + n0b + bn;
  const size_t bStep = (size_t)32 * ldb;

  v8f acc0 = {}, acc1 = {};
  uint2 aReg;
  uint4 bReg0, bReg1;

  auto gload = [&]() {
    aReg = *(const uint2*)aPtr;
    if (FULLN) {
      bReg0 = *(const uint4*)(bPtr);
      bReg1 = *(const uint4*)(bPtr + 8);
    } else {
      bReg0.x = bReg0.y = bReg0.z = bReg0.w = 0u;
      bReg1.x = bReg1.y = bReg1.z = bReg1.w = 0u;
      if (bok0) bReg0 = *(const uint4*)(bPtr);
      if (bok1) bReg1 = *(const uint4*)(bPtr + 8);
    }
    aPtr += 32;
    bPtr += bStep;
  };
  auto lstore = [&](int buf) {
    *(uint2*)&sA[buf][ar * LDA_P + ac]     = aReg;
    *(uint4*)&sB[buf][bk * LDB_P + bn]     = bReg0;
    *(uint4*)&sB[buf][bk * LDB_P + bn + 8] = bReg1;
  };

  gload();
  lstore(0);
  __syncthreads();

  for (int i = 0; i < nk; ++i) {
    const int cur = i & 1;
    if (i + 1 < nk) gload();

    const bf16_t* pa = &sA[cur][arow * LDA_P];
    v8bf alo = *(const v8bf*)(pa + kbase);
    v8bf ahi = *(const v8bf*)(pa + 16 + kbase);
    const bf16_t* pb = &sB[cur][kbase * LDB_P + nt * 32 + (lane & 15)];

    v16bf a, b0, b1;
#pragma unroll
    for (int j = 0; j < 8; ++j) {
      a[j]  = alo[j];  a[j + 8]  = ahi[j];
      b0[j] = pb[j * LDB_P];      b0[j + 8] = pb[(16 + j) * LDB_P];
      b1[j] = pb[j * LDB_P + 16]; b1[j + 8] = pb[(16 + j) * LDB_P + 16];
    }
    acc0 = __builtin_amdgcn_wmma_f32_16x16x32_bf16(false, a, false, b0, (short)0, acc0,
                                                   false, false);
    acc1 = __builtin_amdgcn_wmma_f32_16x16x32_bf16(false, a, false, b1, (short)0, acc1,
                                                   false, false);
    if (i + 1 < nk) lstore(cur ^ 1);
    __syncthreads();
  }

  // C/D layout: VGPR v -> row m0+v (lanes 0-15) or m0+8+v (lanes 16-31);
  // lane&15 -> column.
  const int crow0 = m0b + mt * 16 + ((lane >> 4) << 3);
#pragma unroll
  for (int tn = 0; tn < 2; ++tn) {
    const int ccol = n0b + nt * 32 + tn * 16 + (lane & 15);
    if (!FULLN && ccol >= N) continue;
    v8f accv = tn ? acc1 : acc0;
    const float bv = bias[ccol];
#pragma unroll
    for (int v = 0; v < 8; ++v) {
      const int r = crow0 + v;
      float val = accv[v] + bv;
      if (EP >= 1) val += resid[(size_t)r * ldc + ccol];
      if (EP == 2) val *= mask[(size_t)r * ldc + ccol];
      C[(size_t)r * ldc + ccol] = val;
    }
  }
}

// ---------------- host orchestration ----------------
extern "C" void kernel_launch(void* const* d_in, const int* in_sizes, int n_in,
                              void* d_out, int out_size, void* d_ws, size_t ws_size,
                              hipStream_t stream) {
  (void)in_sizes; (void)n_in; (void)out_size; (void)ws_size;

  const float* input = (const float*)d_in[0];
  const float* Wih0  = (const float*)d_in[1];
  const float* Whh0  = (const float*)d_in[2];
  const float* bih0  = (const float*)d_in[3];
  const float* bhh0  = (const float*)d_in[4];
  const float* Wih1  = (const float*)d_in[5];
  const float* Whh1  = (const float*)d_in[6];
  const float* bih1  = (const float*)d_in[7];
  const float* bhh1  = (const float*)d_in[8];
  const float* Wlin  = (const float*)d_in[9];
  const float* blin  = (const float*)d_in[10];

  float* out_enc = (float*)d_out;
  float* out_dec = out_enc + (size_t)T_LEN * BATCH * FDIM;

  char* p = (char*)d_ws;
  auto take = [&](size_t bytes) { char* r = p; p += (bytes + 255) & ~(size_t)255; return r; };

  bf16_t* xen   = (bf16_t*)take((size_t)T_LEN * BATCH * RNN_IN * 2);
  bf16_t* wih0t = (bf16_t*)take((size_t)G3 * RNN_IN * 2);   // K x N = 288 x 1536
  bf16_t* whh0t = (bf16_t*)take((size_t)G3 * HDIM * 2);     // 512 x 1536
  bf16_t* wih1t = (bf16_t*)take((size_t)G3 * HDIM * 2);
  bf16_t* whh1t = (bf16_t*)take((size_t)G3 * HDIM * 2);
  bf16_t* wlint = (bf16_t*)take((size_t)FDIM * HDIM * 2);   // 512 x 96
  float*  maskm = (float*)take((size_t)BATCH * FDIM * 4);
  float*  gi    = (float*)take((size_t)BATCH * G3 * 4);
  float*  gh    = (float*)take((size_t)BATCH * G3 * 4);
  bf16_t* xdec  = (bf16_t*)take((size_t)BATCH * RNN_IN * 2);
  // hidden states contiguous -> one async memset zeroes f32 and bf16 alike
  char* hbase = take((size_t)BATCH * HDIM * (4 + 4 + 2 + 2));
  float*  h0f = (float*)hbase;
  float*  h1f = h0f + (size_t)BATCH * HDIM;
  bf16_t* h0b = (bf16_t*)(h1f + (size_t)BATCH * HDIM);
  bf16_t* h1b = h0b + (size_t)BATCH * HDIM;

  hipMemsetAsync(hbase, 0, (size_t)BATCH * HDIM * 12, stream);

  const int TB = 256;
  auto cdiv = [](int a, int b) { return (a + b - 1) / b; };

  transpose_to_bf16_k<<<cdiv(G3 * RNN_IN, TB), TB, 0, stream>>>(Wih0, wih0t, G3, RNN_IN);
  transpose_to_bf16_k<<<cdiv(G3 * HDIM, TB), TB, 0, stream>>>(Whh0, whh0t, G3, HDIM);
  transpose_to_bf16_k<<<cdiv(G3 * HDIM, TB), TB, 0, stream>>>(Wih1, wih1t, G3, HDIM);
  transpose_to_bf16_k<<<cdiv(G3 * HDIM, TB), TB, 0, stream>>>(Whh1, whh1t, G3, HDIM);
  transpose_to_bf16_k<<<cdiv(FDIM * HDIM, TB), TB, 0, stream>>>(Wlin, wlint, FDIM, HDIM);
  mask_k<<<cdiv(BATCH * FDIM, TB), TB, 0, stream>>>(input, maskm);
  build_xen_k<<<cdiv(T_LEN * BATCH * FDIM, TB), TB, 0, stream>>>(input, xen);

  dim3 gemm_big(G3 / 128, BATCH / 32);       // 12 x 8 blocks, branch-free
  dim3 gemm_proj(1, BATCH / 32);             // N=96 fits one 128-wide tile
  int ew_bh = cdiv(BATCH * HDIM, TB);
  int ew_bf = cdiv(BATCH * FDIM, TB);

  // ---------------- encoder ----------------
  for (int t = 0; t < T_LEN; ++t) {
    wmma_gemm_bf16<true, 0><<<gemm_big, 256, 0, stream>>>(
        xen + (size_t)t * BATCH * RNN_IN, RNN_IN, wih0t, G3, gi, G3, G3, RNN_IN,
        bih0, nullptr, nullptr);
    wmma_gemm_bf16<true, 0><<<gemm_big, 256, 0, stream>>>(
        h0b, HDIM, whh0t, G3, gh, G3, G3, HDIM, bhh0, nullptr, nullptr);
    gru_combine_k<<<ew_bh, TB, 0, stream>>>(gi, gh, h0f, h0b);
    wmma_gemm_bf16<true, 0><<<gemm_big, 256, 0, stream>>>(
        h0b, HDIM, wih1t, G3, gi, G3, G3, HDIM, bih1, nullptr, nullptr);
    wmma_gemm_bf16<true, 0><<<gemm_big, 256, 0, stream>>>(
        h1b, HDIM, whh1t, G3, gh, G3, G3, HDIM, bhh1, nullptr, nullptr);
    gru_combine_k<<<ew_bh, TB, 0, stream>>>(gi, gh, h1f, h1b);
    // fused projection epilogue: out = h1 @ Wlin^T + b_lin + input[t]
    wmma_gemm_bf16<false, 1><<<gemm_proj, 256, 0, stream>>>(
        h1b, HDIM, wlint, FDIM, out_enc + (size_t)t * BATCH * FDIM, FDIM, FDIM, HDIM,
        blin, input + (size_t)t * BATCH * FDIM, nullptr);
  }

  // ---------------- decoder ----------------
  dec_prep0_k<<<ew_bf, TB, 0, stream>>>(input, xdec);
  for (int t = 0; t < T_LEN; ++t) {
    wmma_gemm_bf16<true, 0><<<gemm_big, 256, 0, stream>>>(
        xdec, RNN_IN, wih0t, G3, gi, G3, G3, RNN_IN, bih0, nullptr, nullptr);
    wmma_gemm_bf16<true, 0><<<gemm_big, 256, 0, stream>>>(
        h0b, HDIM, whh0t, G3, gh, G3, G3, HDIM, bhh0, nullptr, nullptr);
    gru_combine_k<<<ew_bh, TB, 0, stream>>>(gi, gh, h0f, h0b);
    wmma_gemm_bf16<true, 0><<<gemm_big, 256, 0, stream>>>(
        h0b, HDIM, wih1t, G3, gi, G3, G3, HDIM, bih1, nullptr, nullptr);
    wmma_gemm_bf16<true, 0><<<gemm_big, 256, 0, stream>>>(
        h1b, HDIM, whh1t, G3, gh, G3, G3, HDIM, bhh1, nullptr, nullptr);
    gru_combine_k<<<ew_bh, TB, 0, stream>>>(gi, gh, h1f, h1b);
    // fused projection epilogue: pred = (h1 @ Wlin^T + b_lin + prev) * mask
    const float* prev = (t == 0) ? input : out_dec + (size_t)(t - 1) * BATCH * FDIM;
    float* pred = out_dec + (size_t)t * BATCH * FDIM;
    wmma_gemm_bf16<false, 2><<<gemm_proj, 256, 0, stream>>>(
        h1b, HDIM, wlint, FDIM, pred, FDIM, FDIM, HDIM, blin, prev, maskm);
    dec_buildx_k<<<ew_bf, TB, 0, stream>>>(pred, xdec, t);
  }
}
